// GATModel_85409719648817
// MI455X (gfx1250) — matrix operand order repaired and checked by reading, not compiled, not run on hardware
//
#include <hip/hip_runtime.h>
#include <hip/hip_bf16.h>

typedef __attribute__((ext_vector_type(16))) _Float16 v16h;
typedef __attribute__((ext_vector_type(8)))  _Float16 v8h;
typedef __attribute__((ext_vector_type(4)))  _Float16 v4h;
typedef __attribute__((ext_vector_type(8)))  float    v8f;

// ---------- helpers ----------
__device__ __forceinline__ unsigned fenc(float f) {
  unsigned u = __float_as_uint(f);
  return (u & 0x80000000u) ? ~u : (u | 0x80000000u);   // monotonic float->uint
}
__device__ __forceinline__ float fdec(unsigned u) {
  unsigned v = (u & 0x80000000u) ? (u & 0x7FFFFFFFu) : ~u;
  return __uint_as_float(v);
}
__device__ __forceinline__ float elu1(float v) {
  return (v > 0.0f) ? v : (expf(v) - 1.0f);
}

__global__ void fill_u32_kernel(unsigned* p, unsigned v, int n) {
  int i = blockIdx.x * blockDim.x + threadIdx.x;
  if (i < n) p[i] = v;
}

// ---------- self-loop edge attr (fill_value='mean') ----------
__global__ void deg_attr_kernel(const int* __restrict__ ei, const float* __restrict__ eattr,
                                float* deg, float* lsum, int E) {
  int e = blockIdx.x * blockDim.x + threadIdx.x;
  if (e >= E) return;
  int dst = ei[E + e];
  atomicAdd(&deg[dst], 1.0f);
  const float4 a = *(const float4*)&eattr[(long)e * 4];
  atomicAdd(&lsum[dst * 4 + 0], a.x);
  atomicAdd(&lsum[dst * 4 + 1], a.y);
  atomicAdd(&lsum[dst * 4 + 2], a.z);
  atomicAdd(&lsum[dst * 4 + 3], a.w);
}

__global__ void loop_norm_kernel(float* lsum, const float* __restrict__ deg, int N) {
  int n = blockIdx.x * blockDim.x + threadIdx.x;
  if (n >= N) return;
  float d = fmaxf(deg[n], 1.0f);
#pragma unroll
  for (int k = 0; k < 4; ++k) lsum[n * 4 + k] /= d;
}

// ---------- Wsd[k][j]: j<H -> sum_c W[k,j*C+c]*att_src[j,c]; j in [H,2H) -> att_dst; else 0 ----------
__global__ void precompute_attw_kernel(const float* __restrict__ W,
                                       const float* __restrict__ attS, const float* __restrict__ attD,
                                       float* __restrict__ Wsd, int heads, int chead) {
  int k = threadIdx.x;            // 0..127
  if (k >= 128) return;
  for (int j = 0; j < 16; ++j) {
    float s = 0.0f;
    if (j < heads) {
      for (int c = 0; c < chead; ++c) s += W[k * 128 + j * chead + c] * attS[j * chead + c];
    } else if (j < 2 * heads) {
      int h = j - heads;
      for (int c = 0; c < chead; ++c) s += W[k * 128 + h * chead + c] * attD[h * chead + c];
    }
    Wsd[k * 16 + j] = s;
  }
}

// ---------- u[k,h] = sum_c W_e[k, h*C+c] * att_edge[h,c] ----------
__global__ void precompute_u_kernel(const float* __restrict__ We, const float* __restrict__ attE,
                                    float* u, int heads, int chead) {
  int h = threadIdx.x;
  if (h >= heads) return;
  for (int k = 0; k < 4; ++k) {
    float s = 0.0f;
    for (int c = 0; c < chead; ++c) s += We[k * 128 + h * chead + c] * attE[h * chead + c];
    u[k * heads + h] = s;
  }
}

// ---------- node linear: H = X @ W (N x 128 x 128) + fused [a_src|a_dst] = X @ Wsd ----------
// 64 rows (4 M-tiles) per block; 8 waves = 8 N-tiles; B fragments hoisted in registers.
// Waves 0-3 additionally compute the 16-wide attention tile (aSD[n][0..H-1]=a_src,
// [H..2H-1]=a_dst). Output buffers are padded to 64-row multiples, so all stores are
// branchless immediate-offset clause stores; only the X loads are bounds-guarded.
template <int HEADS>
__global__ __launch_bounds__(256) void node_linear_wmma_kernel(
    const float* __restrict__ X, const float* __restrict__ W, const float* __restrict__ Wsd,
    float* __restrict__ H, float* __restrict__ aSD, int N)
{
  __shared__ _Float16 Xs[64 * 128];    // 64-row tile of X (row-major, f16)
  __shared__ _Float16 Wt[128 * 128];   // W transposed: Wt[n][k] (f16)
  __shared__ _Float16 At[16 * 128];    // Wsd transposed: At[j][k] (f16)
  const int tid = threadIdx.x;
  const int m0  = blockIdx.x * 64;

  // W -> LDS (transposed): coalesced float4 global reads, scalar LDS writes
  for (int i = tid * 4; i < 128 * 128; i += 256 * 4) {
    int k = i >> 7, c = i & 127;
    const float4 w4 = *(const float4*)&W[i];
    Wt[(c + 0) * 128 + k] = (_Float16)w4.x;
    Wt[(c + 1) * 128 + k] = (_Float16)w4.y;
    Wt[(c + 2) * 128 + k] = (_Float16)w4.z;
    Wt[(c + 3) * 128 + k] = (_Float16)w4.w;
  }
  // Wsd -> LDS (transposed)
  for (int i = tid; i < 128 * 16; i += 256) {
    int k = i >> 4, j = i & 15;
    At[j * 128 + k] = (_Float16)Wsd[i];
  }
  // X tile -> LDS: float4 reads (zero-fill past N), v4h writes
  for (int i = tid * 4; i < 64 * 128; i += 256 * 4) {
    int m = i >> 7, k = i & 127;
    int row = m0 + m;
    v4h h4 = {};
    if (row < N) {
      const float4 x4 = *(const float4*)&X[(long)row * 128 + k];
      h4[0] = (_Float16)x4.x; h4[1] = (_Float16)x4.y;
      h4[2] = (_Float16)x4.z; h4[3] = (_Float16)x4.w;
    }
    *(v4h*)&Xs[i] = h4;
  }
  __syncthreads();

  const int wave = tid >> 5;     // 8 waves -> 8 column tiles of 16
  const int lane = tid & 31;
  const int half = lane >> 4;
  const int ml   = lane & 15;
  const int n0   = wave * 16;

  // B fragments (row-tile invariant): lane%16 = col N, half selects contiguous K 16-block
  v16h bfrag[4];
#pragma unroll
  for (int kt = 0; kt < 4; ++kt)
    bfrag[kt] = *(const v16h*)&Wt[(n0 + ml) * 128 + kt * 32 + half * 16];

  const int col = n0 + ml;       // 0..127

#pragma unroll
  for (int t = 0; t < 4; ++t) {
    const int mb = t * 16;
    v8f acc = {};
#pragma unroll
    for (int kt = 0; kt < 4; ++kt) {
      // A 16x32 f16: lane(0-15)=row M, half selects contiguous K 16-block (0-15 | 16-31)
      const v8h alo = *(const v8h*)&Xs[(mb + ml) * 128 + kt * 32 + half * 16];
      const v8h ahi = *(const v8h*)&Xs[(mb + ml) * 128 + kt * 32 + half * 16 + 8];
      const v16h a = __builtin_shufflevector(alo, ahi, 0, 1, 2, 3, 4, 5, 6, 7,
                                             8, 9, 10, 11, 12, 13, 14, 15);
      acc = __builtin_amdgcn_wmma_f32_16x16x32_f16(false, a, false, bfrag[kt],
                                                   (short)0, acc, false, false);
    }
    // D layout: lane L, vgpr r -> row M = r + 8*(L/16), col N = L%16
    float* hp = &H[(long)(m0 + mb + half * 8) * 128 + col];
#pragma unroll
    for (int r = 0; r < 8; ++r) hp[(long)r * 128] = acc[r];   // imm-offset clause stores
  }

  // attention tile: waves 0-3 handle row tiles 0-3 (wave-uniform branch; EXEC stays all-1s)
  if (wave < 4) {
    const int mb = wave * 16;
    v8f acc = {};
#pragma unroll
    for (int kt = 0; kt < 4; ++kt) {
      const v8h alo = *(const v8h*)&Xs[(mb + ml) * 128 + kt * 32 + half * 16];
      const v8h ahi = *(const v8h*)&Xs[(mb + ml) * 128 + kt * 32 + half * 16 + 8];
      const v16h a = __builtin_shufflevector(alo, ahi, 0, 1, 2, 3, 4, 5, 6, 7,
                                             8, 9, 10, 11, 12, 13, 14, 15);
      const v16h b = *(const v16h*)&At[ml * 128 + kt * 32 + half * 16];
      acc = __builtin_amdgcn_wmma_f32_16x16x32_f16(false, a, false, b,
                                                   (short)0, acc, false, false);
    }
    const int r0 = m0 + mb + half * 8;
    float* ap = &aSD[(long)r0 * (2 * HEADS) + ml];
    if (ml < 2 * HEADS) {              // always true for HEADS==8 (folds away)
#pragma unroll
      for (int r = 0; r < 8; ++r) ap[(long)r * 2 * HEADS] = acc[r];
    }
  }
}

// ---------- alpha = leaky_relu(a_src[src]+a_dst[dst]+a_e); segment max over dst ----------
template <int HEADS>
__global__ void edge_alpha_kernel(const int* __restrict__ ei, const float* __restrict__ eattr,
                                  const float* __restrict__ loopA,
                                  const float* __restrict__ aSD, const float* __restrict__ u,
                                  float* __restrict__ alpha, unsigned* __restrict__ amax,
                                  int E, int EF) {
  int e = blockIdx.x * blockDim.x + threadIdx.x;
  if (e >= EF) return;
  int src, dst; float4 a;
  if (e < E) {
    src = ei[e]; dst = ei[E + e];
    a = *(const float4*)&eattr[(long)e * 4];
  } else {
    int n = e - E; src = n; dst = n;
    a = *(const float4*)&loopA[(long)n * 4];
  }
#pragma unroll
  for (int h = 0; h < HEADS; ++h) {
    float ae = a.x * u[0 * HEADS + h] + a.y * u[1 * HEADS + h] +
               a.z * u[2 * HEADS + h] + a.w * u[3 * HEADS + h];
    float al = aSD[src * 2 * HEADS + h] + aSD[dst * 2 * HEADS + HEADS + h] + ae;
    al = (al > 0.0f) ? al : 0.2f * al;
    alpha[(long)e * HEADS + h] = al;
    atomicMax(&amax[dst * HEADS + h], fenc(al));
  }
}

// ---------- ex = exp(alpha - amax[dst]); denom[dst] += ex ----------
template <int HEADS>
__global__ void edge_exp_kernel(const int* __restrict__ ei, float* __restrict__ alpha,
                                const unsigned* __restrict__ amax, float* __restrict__ denom,
                                int E, int EF) {
  long idx = (long)blockIdx.x * blockDim.x + threadIdx.x;
  if (idx >= (long)EF * HEADS) return;
  int e = (int)(idx / HEADS);
  int h = (int)(idx % HEADS);
  int dst = (e < E) ? ei[E + e] : (e - E);
  float ex = expf(alpha[idx] - fdec(amax[dst * HEADS + h]));
  alpha[idx] = ex;
  atomicAdd(&denom[dst * HEADS + h], ex);
}

// ---------- agg[dst, col] += H[src, col] * ex/(denom[dst]+eps), 4 cols per thread ----------
template <int HEADS>
__global__ void edge_agg_kernel(const int* __restrict__ ei, const float* __restrict__ ex,
                                const float* __restrict__ denom, const float* __restrict__ H,
                                float* __restrict__ agg, int E, int EF) {
  constexpr int CHEAD = 128 / HEADS;
  long idx = (long)blockIdx.x * blockDim.x + threadIdx.x;
  if (idx >= (long)EF * 32) return;
  int col = (int)(idx & 31) * 4;       // 4-col group stays inside one head (CHEAD>=16)
  int e   = (int)(idx >> 5);
  int h   = col / CHEAD;
  int src, dst;
  if (e < E) { src = ei[e]; dst = ei[E + e]; } else { src = e - E; dst = src; }
  float w = ex[(long)e * HEADS + h] / (denom[dst * HEADS + h] + 1e-16f);
  const float4 hv = *(const float4*)&H[(long)src * 128 + col];
  float* ap = &agg[(long)dst * 128 + col];
  atomicAdd(ap + 0, hv.x * w);
  atomicAdd(ap + 1, hv.y * w);
  atomicAdd(ap + 2, hv.z * w);
  atomicAdd(ap + 3, hv.w * w);
}

// ---------- out = elu(agg + bias), 4 cols per thread ----------
__global__ void node_finalize_kernel(const float* __restrict__ agg, const float* __restrict__ bias,
                                     float* __restrict__ out, int N) {
  long idx = (long)blockIdx.x * blockDim.x + threadIdx.x;
  if (idx >= (long)N * 32) return;
  int c4 = (int)(idx & 31) * 4;
  const float4 a = *(const float4*)&agg[idx * 4];
  const float4 b = *(const float4*)&bias[c4];
  float4 o;
  o.x = elu1(a.x + b.x); o.y = elu1(a.y + b.y);
  o.z = elu1(a.z + b.z); o.w = elu1(a.w + b.w);
  *(float4*)&out[idx * 4] = o;
}

// ---------- global mean pool (scatter phase), 4 cols per thread ----------
__global__ void pool_scatter_kernel(const float* __restrict__ h, const int* __restrict__ batch,
                                    float* pooled, float* counts, int N) {
  long idx = (long)blockIdx.x * blockDim.x + threadIdx.x;
  if (idx >= (long)N * 32) return;
  int n = (int)(idx >> 5), q = (int)(idx & 31);
  int g = batch[n];
  const float4 hv = *(const float4*)&h[(long)n * 128 + q * 4];
  float* pp = &pooled[(long)g * 128 + q * 4];
  atomicAdd(pp + 0, hv.x);
  atomicAdd(pp + 1, hv.y);
  atomicAdd(pp + 2, hv.z);
  atomicAdd(pp + 3, hv.w);
  if (q == 0) atomicAdd(&counts[g], 1.0f);
}

// ---------- per-graph MLP: sigmoid(fc2(relu(fc1(mean)))) ----------
__global__ void mlp_kernel(const float* __restrict__ pooled, const float* __restrict__ counts,
                           const float* __restrict__ fc1w, const float* __restrict__ fc1b,
                           const float* __restrict__ fc2w, const float* __restrict__ fc2b,
                           float* __restrict__ out, int G) {
  __shared__ float pl[128];
  __shared__ float red[64];
  int g = blockIdx.x;
  int j = threadIdx.x;                  // 0..63
  float cnt = fmaxf(counts[g], 1.0f);
  pl[j]      = pooled[(long)g * 128 + j] / cnt;
  pl[j + 64] = pooled[(long)g * 128 + j + 64] / cnt;
  __syncthreads();
  float acc = fc1b[j];
  for (int k = 0; k < 128; ++k) acc += pl[k] * fc1w[k * 64 + j];
  float hid = fmaxf(acc, 0.0f);
  red[j] = hid * fc2w[j];
  __syncthreads();
  for (int s = 32; s > 0; s >>= 1) {
    if (j < s) red[j] += red[j + s];
    __syncthreads();
  }
  if (j == 0) out[g] = 1.0f / (1.0f + expf(-(red[0] + fc2b[0])));
}

// ---------- host-side layer driver ----------
struct ConvP { const float *W, *We, *asrc, *adst, *aedge, *bias; };
struct LayerWs {
  float *bufH, *bufAgg, *aSD, *denom, *alpha, *loopA, *uvec, *Wsd;
  unsigned* aMax;
};

static inline int cdiv_l(long a, long b) { return (int)((a + b - 1) / b); }

template <int HEADS>
static void run_gat_layer(const float* xin, const ConvP& cp, float* xout, const LayerWs& w,
                          const int* ei, const float* eattr, int N, int E, int EF,
                          hipStream_t stream) {
  constexpr int CHEAD = 128 / HEADS;
  const int TB = 256;
  precompute_attw_kernel<<<1, 128, 0, stream>>>(cp.W, cp.asrc, cp.adst, w.Wsd, HEADS, CHEAD);
  precompute_u_kernel<<<1, 32, 0, stream>>>(cp.We, cp.aedge, w.uvec, HEADS, CHEAD);
  node_linear_wmma_kernel<HEADS><<<cdiv_l(N, 64), 256, 0, stream>>>(
      xin, cp.W, w.Wsd, w.bufH, w.aSD, N);
  fill_u32_kernel<<<cdiv_l((long)N * HEADS, TB), TB, 0, stream>>>(w.aMax, 0x007FFFFFu, N * HEADS);
  edge_alpha_kernel<HEADS><<<cdiv_l(EF, TB), TB, 0, stream>>>(
      ei, eattr, w.loopA, w.aSD, w.uvec, w.alpha, w.aMax, E, EF);
  hipMemsetAsync(w.denom, 0, (size_t)N * HEADS * 4, stream);
  edge_exp_kernel<HEADS><<<cdiv_l((long)EF * HEADS, TB), TB, 0, stream>>>(
      ei, w.alpha, w.aMax, w.denom, E, EF);
  hipMemsetAsync(w.bufAgg, 0, (size_t)N * 128 * 4, stream);
  edge_agg_kernel<HEADS><<<cdiv_l((long)EF * 32, TB), TB, 0, stream>>>(
      ei, w.alpha, w.denom, w.bufH, w.bufAgg, E, EF);
  node_finalize_kernel<<<cdiv_l((long)N * 32, TB), TB, 0, stream>>>(w.bufAgg, cp.bias, xout, N);
}

extern "C" void kernel_launch(void* const* d_in, const int* in_sizes, int n_in,
                              void* d_out, int out_size, void* d_ws, size_t ws_size,
                              hipStream_t stream) {
  (void)n_in; (void)ws_size;
  const float* x0    = (const float*)d_in[0];
  const int*   ei    = (const int*)d_in[1];
  const float* eattr = (const float*)d_in[2];
  const int*   batch = (const int*)d_in[3];
  const int N  = in_sizes[0] / 128;
  const int E  = in_sizes[1] / 2;
  const int EF = E + N;
  const int G  = out_size;
  const int NP = (N + 63) & ~63;       // padded row count (branchless WMMA stores)

  auto getConv = [&](int b) {
    ConvP p;
    p.W = (const float*)d_in[b]; p.We = (const float*)d_in[b + 1];
    p.asrc = (const float*)d_in[b + 2]; p.adst = (const float*)d_in[b + 3];
    p.aedge = (const float*)d_in[b + 4]; p.bias = (const float*)d_in[b + 5];
    return p;
  };
  ConvP c1 = getConv(4), c2 = getConv(10), c3 = getConv(16);
  const float* fc1w = (const float*)d_in[22];
  const float* fc1b = (const float*)d_in[23];
  const float* fc2w = (const float*)d_in[24];
  const float* fc2b = (const float*)d_in[25];

  // workspace carve (256B aligned)
  char* wp = (char*)d_ws;
  auto carve = [&](size_t bytes) -> void* {
    void* r = (void*)wp;
    wp += (bytes + 255) & ~(size_t)255;
    return r;
  };
  float* bufX = (float*)carve((size_t)NP * 128 * 4);  // current features (padded)
  LayerWs w;
  w.bufH   = (float*)carve((size_t)NP * 128 * 4);     // padded
  w.bufAgg = (float*)carve((size_t)NP * 128 * 4);     // padded (agg only uses N rows)
  w.aSD    = (float*)carve((size_t)NP * 16 * 4);      // [a_src | a_dst] interleaved (padded)
  w.aMax   = (unsigned*)carve((size_t)N * 8 * 4);
  w.denom  = (float*)carve((size_t)N * 8 * 4);
  w.alpha  = (float*)carve((size_t)EF * 8 * 4);
  w.loopA  = (float*)carve((size_t)N * 4 * 4);
  float* deg = (float*)carve((size_t)N * 4);
  w.uvec   = (float*)carve(256);
  w.Wsd    = (float*)carve((size_t)128 * 16 * 4);
  float* pooled = (float*)carve((size_t)G * 128 * 4);
  float* counts = (float*)carve((size_t)G * 4);

  const int TB = 256;

  // self-loop edge attrs
  hipMemsetAsync(deg, 0, (size_t)N * 4, stream);
  hipMemsetAsync(w.loopA, 0, (size_t)N * 4 * 4, stream);
  deg_attr_kernel<<<cdiv_l(E, TB), TB, 0, stream>>>(ei, eattr, deg, w.loopA, E);
  loop_norm_kernel<<<cdiv_l(N, TB), TB, 0, stream>>>(w.loopA, deg, N);

  run_gat_layer<8>(x0,   c1, bufX, w, ei, eattr, N, E, EF, stream);
  run_gat_layer<8>(bufX, c2, bufX, w, ei, eattr, N, E, EF, stream);
  run_gat_layer<1>(bufX, c3, bufX, w, ei, eattr, N, E, EF, stream);

  hipMemsetAsync(pooled, 0, (size_t)G * 128 * 4, stream);
  hipMemsetAsync(counts, 0, (size_t)G * 4, stream);
  pool_scatter_kernel<<<cdiv_l((long)N * 32, TB), TB, 0, stream>>>(bufX, batch, pooled, counts, N);
  mlp_kernel<<<G, 64, 0, stream>>>(pooled, counts, fc1w, fc1b, fc2w, fc2b, (float*)d_out, G);
}
